// EuclideanFastAttention_54674933678518
// MI455X (gfx1250) — matrix operand order repaired and checked by reading, not compiled, not run on hardware
//
#include <hip/hip_runtime.h>
#include <hip/hip_bf16.h>

// Problem constants (from reference)
#define N_  2048
#define B_  8
#define S_  4
#define F_  64
#define J_  32
#define G_  14
#define M_  256   // N_/B_
#define D_  256   // P*S*F

typedef __attribute__((ext_vector_type(16))) __bf16          v16bf;
typedef __attribute__((ext_vector_type(8)))  float           v8f;

static __device__ __forceinline__ unsigned short f2bf(float f) {
    unsigned u = __builtin_bit_cast(unsigned, f);
    u += 0x7FFFu + ((u >> 16) & 1u);   // round-to-nearest-even
    return (unsigned short)(u >> 16);
}

static __device__ __forceinline__ v8f wmma_bf16(v16bf a, v16bf b, v8f c) {
    return __builtin_amdgcn_wmma_f32_16x16x32_bf16(
        /*neg_a=*/false, a, /*neg_b=*/false, b,
        /*c_mod=*/(short)0, c, /*reuse_a=*/false, /*reuse_b=*/false);
}

struct frag128 { uint4 lo, hi; };   // 32B == v16bf

// A fragment (16x32 bf16) from LDS panel stored [row][k], k contiguous.
// Lane layout (ISA 7.12.2): lanes 0-15 row=lane, K={0..7,16..23};
// lanes 16-31 row=lane-16, K={8..15,24..31}.
static __device__ __forceinline__ v16bf frag_A_vec(const unsigned short* rowbase, int half) {
    frag128 f;
    f.lo = *(const uint4*)(rowbase + 8 * half);
    f.hi = *(const uint4*)(rowbase + 16 + 8 * half);
    return __builtin_bit_cast(v16bf, f);
}

// B fragment (32x16 bf16) from LDS panel stored [col][k], k contiguous.
// Lanes 0-15: col=lane, K={0..7,16..23}; lanes 16-31: col=lane-16, K={8..15,24..31}.
static __device__ __forceinline__ v16bf frag_B_vec(const unsigned short* colbase, int half) {
    frag128 f;
    f.lo = *(const uint4*)(colbase + 8 * half);
    f.hi = *(const uint4*)(colbase + 16 + 8 * half);
    return __builtin_bit_cast(v16bf, f);
}

// gfx1250 async global->LDS 16-byte copy (VGLOBAL encoding, ASYNCcnt).
static __device__ __forceinline__ void async_copy16(unsigned short* lds_dst,
                                                    const unsigned short* gsrc) {
    unsigned l = (unsigned)(uintptr_t)lds_dst;                 // LDS offset (addr[31:0])
    unsigned long long g = (unsigned long long)(uintptr_t)gsrc;
    asm volatile("global_load_async_to_lds_b128 %0, %1, off"
                 :: "v"(l), "v"(g) : "memory");
}
static __device__ __forceinline__ void wait_async() {
    asm volatile("s_wait_asynccnt 0" ::: "memory");
}

// ---------------------------------------------------------------------------
// Kernel 1: RoPE rotation + pack to bf16.
//   q_bf [b,g,m,d]  rotated features (K-innermost for GEMM-2 A panels)
//   vT_bf[b,e,m]    inputs * node_mask, transposed (K=m innermost, GEMM-1 B)
// ---------------------------------------------------------------------------
__global__ void rope_pack_kernel(const float* __restrict__ inputs,
                                 const float* __restrict__ positions,
                                 const int* __restrict__ segs,
                                 const unsigned char* __restrict__ gmask,
                                 const float* __restrict__ freqs,
                                 const float* __restrict__ grid_u,
                                 unsigned short* __restrict__ q_bf,
                                 unsigned short* __restrict__ vT_bf) {
    int blk = blockIdx.x;
    int n = blk / G_, g = blk % G_;
    int t = threadIdx.x;          // 0..127
    int s = t >> 5, j = t & 31;

    float px = positions[n * 3 + 0];
    float py = positions[n * 3 + 1];
    float pz = positions[n * 3 + 2];
    float dot = px * grid_u[g * 3 + 0] + py * grid_u[g * 3 + 1] + pz * grid_u[g * 3 + 2];
    float ang = dot * freqs[j];
    float c = cosf(ang), sn = sinf(ang);

    int b = n / M_, m = n % M_;
    size_t in_off = ((size_t)n * S_ + s) * F_ + 2 * j;
    float x1 = inputs[in_off];
    float x2 = inputs[in_off + 1];
    float r1 = x1 * c - x2 * sn;
    float r2 = x1 * sn + x2 * c;

    int d = s * F_ + 2 * j;
    size_t qoff = ((size_t)((b * G_ + g) * M_ + m)) * D_ + d;
    q_bf[qoff]     = f2bf(r1);
    q_bf[qoff + 1] = f2bf(r2);

    if (g == 0) {
        float mk = gmask[segs[n]] ? 1.0f : 0.0f;
        size_t v0 = ((size_t)(b * D_ + d)) * M_ + m;        // [b][e][m]
        vT_bf[v0]      = f2bf(x1 * mk);
        vT_bf[v0 + M_] = f2bf(x2 * mk);                     // e = d+1
    }
}

// ---------------------------------------------------------------------------
// Kernel 2: kvT_bf[b,g,e,d] = w_g * sum_m q_bf[b,g,m,d] * v[b,m,e]
// Block tile 64(d) x 128(e); 8 waves each own 32x32 (2x2 WMMA).
// Double-buffered, software-pipelined staging:
//   lA[buf][d_local][m] (global b128 -> regs -> ds_store transpose)
//   lB[buf][e_local][m] (async global->LDS from vT)
// grid = B*G*8 blocks of 256 threads.
// ---------------------------------------------------------------------------
#define LP_STRIDE 40    // 32 k + 8 pad (ushorts per panel row; 80B, 16B-aligned)
#define NSTEP2   (M_ / 32)

__global__ __launch_bounds__(256) void gemm_kv_kernel(
        const unsigned short* __restrict__ q_bf,
        const unsigned short* __restrict__ vT_bf,
        const float* __restrict__ grid_w,
        unsigned short* __restrict__ kvT_bf) {
    __shared__ alignas(16) unsigned short lA[2][64 * LP_STRIDE];   // [d_local][k=m]
    __shared__ alignas(16) unsigned short lB[2][128 * LP_STRIDE];  // [e_local][k=m]

    int blk = blockIdx.x;
    int bg = blk >> 3, t8 = blk & 7;
    int b = bg / G_, g = bg % G_;
    int td = (t8 & 3) * 64;    // d tile base
    int te = (t8 >> 2) * 128;  // e tile base

    int tid = threadIdx.x;
    int wave = tid >> 5, lane = tid & 31;
    int wd = (wave & 1) * 32;
    int we = (wave >> 1) * 32;
    int half = lane >> 4, lrow = lane & 15;

    const unsigned short* qg  = q_bf  + (size_t)(b * G_ + g) * M_ * D_;
    const unsigned short* vTb = vT_bf + (size_t)b * D_ * M_;

    // staging coordinates
    int ar  = tid >> 3;            // A: m row 0..31
    int ac8 = (tid & 7) * 8;       // A: d chunk
    int er0 = tid >> 2, ec0 = (tid & 3) * 8;                 // B chunk 0
    int er1 = (tid + 256) >> 2, ec1 = ((tid + 256) & 3) * 8; // B chunk 1

    v8f acc[2][2] = {};

    // ---- preamble: stage step 0 into buffer 0 ----
    uint4 aval = *(const uint4*)(qg + (size_t)ar * D_ + td + ac8);
    async_copy16(&lB[0][er0 * LP_STRIDE + ec0], vTb + (size_t)(te + er0) * M_ + ec0);
    async_copy16(&lB[0][er1 * LP_STRIDE + ec1], vTb + (size_t)(te + er1) * M_ + ec1);
    {
        unsigned vv[4] = {aval.x, aval.y, aval.z, aval.w};
#pragma unroll
        for (int i = 0; i < 4; ++i) {
            lA[0][(ac8 + 2 * i + 0) * LP_STRIDE + ar] = (unsigned short)(vv[i] & 0xffffu);
            lA[0][(ac8 + 2 * i + 1) * LP_STRIDE + ar] = (unsigned short)(vv[i] >> 16);
        }
    }
    wait_async();
    __syncthreads();

    for (int i = 0; i < NSTEP2; ++i) {
        int cur = i & 1, nxt = cur ^ 1;
        uint4 anext;
        if (i + 1 < NSTEP2) {
            int k1 = (i + 1) * 32;
            // issue next A global load + next B async copies (overlap compute)
            anext = *(const uint4*)(qg + (size_t)(k1 + ar) * D_ + td + ac8);
            async_copy16(&lB[nxt][er0 * LP_STRIDE + ec0],
                         vTb + (size_t)(te + er0) * M_ + k1 + ec0);
            async_copy16(&lB[nxt][er1 * LP_STRIDE + ec1],
                         vTb + (size_t)(te + er1) * M_ + k1 + ec1);
        }

        v16bf a0 = frag_A_vec(&lA[cur][(wd + 0  + lrow) * LP_STRIDE], half);
        v16bf a1 = frag_A_vec(&lA[cur][(wd + 16 + lrow) * LP_STRIDE], half);
        v16bf b0 = frag_B_vec(&lB[cur][(we + 0  + lrow) * LP_STRIDE], half);
        v16bf b1 = frag_B_vec(&lB[cur][(we + 16 + lrow) * LP_STRIDE], half);

        acc[0][0] = wmma_bf16(a0, b0, acc[0][0]);
        acc[0][1] = wmma_bf16(a0, b1, acc[0][1]);
        acc[1][0] = wmma_bf16(a1, b0, acc[1][0]);
        acc[1][1] = wmma_bf16(a1, b1, acc[1][1]);

        if (i + 1 < NSTEP2) {   // transpose-store next A after compute (hides load latency)
            unsigned vv[4] = {anext.x, anext.y, anext.z, anext.w};
#pragma unroll
            for (int k = 0; k < 4; ++k) {
                lA[nxt][(ac8 + 2 * k + 0) * LP_STRIDE + ar] = (unsigned short)(vv[k] & 0xffffu);
                lA[nxt][(ac8 + 2 * k + 1) * LP_STRIDE + ar] = (unsigned short)(vv[k] >> 16);
            }
        }
        wait_async();
        __syncthreads();
    }

    float w = grid_w[g];
    unsigned short* kvTg = kvT_bf + (size_t)(b * G_ + g) * D_ * D_;
#pragma unroll
    for (int di = 0; di < 2; ++di)
#pragma unroll
        for (int ei = 0; ei < 2; ++ei)
#pragma unroll
            for (int r = 0; r < 8; ++r) {
                int drow = td + wd + di * 16 + r + 8 * half;
                int ecol = te + we + ei * 16 + lrow;
                kvTg[(size_t)ecol * D_ + drow] = f2bf(acc[di][ei][r] * w);  // [e][d]
            }
}

// ---------------------------------------------------------------------------
// Kernel 3: out[n=b*M+m, e] = mask_n * sum_g sum_d q_bf[b,g,m,d]*kv[b,g,d,e]
// Flattened 112-step pipelined loop over (g, k0); all panels async-staged:
//   lA[buf][m_local][d] from q_bf, lB[buf][e_local][d] from kvT_bf.
// grid = B*8 blocks of 256 threads.
// ---------------------------------------------------------------------------
#define NSTEP3 (G_ * (D_ / 32))   // 112

__global__ __launch_bounds__(256) void gemm_out_kernel(
        const unsigned short* __restrict__ q_bf,
        const unsigned short* __restrict__ kvT_bf,
        const int* __restrict__ segs,
        const unsigned char* __restrict__ gmask,
        float* __restrict__ out) {
    __shared__ alignas(16) unsigned short lA[2][64 * LP_STRIDE];   // [m_local][k=d]
    __shared__ alignas(16) unsigned short lB[2][128 * LP_STRIDE];  // [e_local][k=d]

    int blk = blockIdx.x;
    int b = blk >> 3, t8 = blk & 7;
    int tm = (t8 & 3) * 64;    // m tile base
    int te = (t8 >> 2) * 128;  // e tile base

    int tid = threadIdx.x;
    int wave = tid >> 5, lane = tid & 31;
    int wm = (wave & 1) * 32;
    int we = (wave >> 1) * 32;
    int half = lane >> 4, lrow = lane & 15;

    // staging coordinates
    int ar = tid >> 2, ac8 = (tid & 3) * 8;                  // A chunk
    int er0 = tid >> 2, ec0 = (tid & 3) * 8;                 // B chunk 0
    int er1 = (tid + 256) >> 2, ec1 = ((tid + 256) & 3) * 8; // B chunk 1

    v8f acc[2][2] = {};

    const size_t qbase  = (size_t)b * G_ * M_ * D_;
    const size_t kvbase = (size_t)b * G_ * D_ * D_;

    // stage(step, buf): g = step>>3, k0 = (step&7)*32
    auto stage = [&](int step, int buf) {
        int g = step >> 3, k0 = (step & 7) * 32;
        const unsigned short* qg   = q_bf   + qbase  + (size_t)g * M_ * D_;
        const unsigned short* kvTg = kvT_bf + kvbase + (size_t)g * D_ * D_;
        async_copy16(&lA[buf][ar * LP_STRIDE + ac8],
                     qg + (size_t)(tm + ar) * D_ + k0 + ac8);
        async_copy16(&lB[buf][er0 * LP_STRIDE + ec0],
                     kvTg + (size_t)(te + er0) * D_ + k0 + ec0);
        async_copy16(&lB[buf][er1 * LP_STRIDE + ec1],
                     kvTg + (size_t)(te + er1) * D_ + k0 + ec1);
    };

    stage(0, 0);
    wait_async();
    __syncthreads();

    for (int i = 0; i < NSTEP3; ++i) {
        int cur = i & 1, nxt = cur ^ 1;
        if (i + 1 < NSTEP3) stage(i + 1, nxt);   // overlap with compute below

        v16bf a0 = frag_A_vec(&lA[cur][(wm + 0  + lrow) * LP_STRIDE], half);
        v16bf a1 = frag_A_vec(&lA[cur][(wm + 16 + lrow) * LP_STRIDE], half);
        v16bf b0 = frag_B_vec(&lB[cur][(we + 0  + lrow) * LP_STRIDE], half);
        v16bf b1 = frag_B_vec(&lB[cur][(we + 16 + lrow) * LP_STRIDE], half);

        acc[0][0] = wmma_bf16(a0, b0, acc[0][0]);
        acc[0][1] = wmma_bf16(a0, b1, acc[0][1]);
        acc[1][0] = wmma_bf16(a1, b0, acc[1][0]);
        acc[1][1] = wmma_bf16(a1, b1, acc[1][1]);

        wait_async();
        __syncthreads();
    }

#pragma unroll
    for (int di = 0; di < 2; ++di)
#pragma unroll
        for (int ei = 0; ei < 2; ++ei)
#pragma unroll
            for (int r = 0; r < 8; ++r) {
                int mrow = tm + wm + di * 16 + r + 8 * half;
                int n = b * M_ + mrow;
                float mk = gmask[segs[n]] ? 1.0f : 0.0f;
                int ecol = te + we + ei * 16 + lrow;
                out[(size_t)n * D_ + ecol] = acc[di][ei][r] * mk;
            }
}

// ---------------------------------------------------------------------------
extern "C" void kernel_launch(void* const* d_in, const int* in_sizes, int n_in,
                              void* d_out, int out_size, void* d_ws, size_t ws_size,
                              hipStream_t stream) {
    const float* inputs          = (const float*)d_in[0];
    const float* positions       = (const float*)d_in[1];
    const int* batch_segments    = (const int*)d_in[2];
    const unsigned char* gmask   = (const unsigned char*)d_in[3]; // bool (1B/elem)
    const float* frequencies     = (const float*)d_in[4];
    const float* grid_u          = (const float*)d_in[5];
    const float* grid_w          = (const float*)d_in[6];
    float* out                   = (float*)d_out;

    // Workspace (bf16 as ushort):
    //   q_bf  : B*G*M*D  (14,680,064 B)
    //   vT_bf : B*D*M    ( 1,048,576 B)
    //   kvT_bf: B*G*D*D  (14,680,064 B)   total ~30.4 MB
    unsigned short* q_bf   = (unsigned short*)d_ws;
    unsigned short* vT_bf  = q_bf  + (size_t)B_ * G_ * M_ * D_;
    unsigned short* kvT_bf = vT_bf + (size_t)B_ * D_ * M_;

    rope_pack_kernel<<<N_ * G_, 128, 0, stream>>>(
        inputs, positions, batch_segments, gmask, frequencies, grid_u, q_bf, vT_bf);

    gemm_kv_kernel<<<B_ * G_ * 8, 256, 0, stream>>>(q_bf, vT_bf, grid_w, kvT_bf);

    gemm_out_kernel<<<B_ * 8, 256, 0, stream>>>(q_bf, kvT_bf, batch_segments, gmask, out);
}
